// LinearCDE_76836964925652
// MI455X (gfx1250) — matrix operand order repaired and checked by reading, not compiled, not run on hardware
//
#include <hip/hip_runtime.h>

// ---------------- problem constants ----------------
#define BB 64     // batch
#define SS 2048   // seq len
#define DD 64     // input dim
#define HH 256    // hidden
#define DP 65     // D+1
#define DPAD 80   // padded d per h -> 5 tiles of 16
#define KY 288    // padded K: 256 y + 1 const(=W_B) + 31 zero
#define KSTEPS (KY / 32)          // 9 wmma k-steps
#define NTILES (HH * DPAD / 16)   // 1280 n-tiles
#define NWG 64
#define TPB 256                   // 8 waves of 32
#define HPW 4                     // h's per WG

// LDS partition (bytes)
#define WLDS_BYTES (HPW * 5 * KSTEPS * 1024)        // 184320: this WG's W slice
#define UU_FLOATS  (BB * DPAD)                      // 5120 per buffer
#define UU_BYTES   (2 * UU_FLOATS * 4)              // 40960 (double buffered)
#define YSL_BYTES  (BB * HPW * 4)                   // 1024
#define LDS_BYTES  (WLDS_BYTES + UU_BYTES + YSL_BYTES)  // 226304 < 320K

typedef __attribute__((ext_vector_type(16))) __bf16 v16bf;
typedef __attribute__((ext_vector_type(8)))  __bf16 v8bf;
typedef __attribute__((ext_vector_type(8)))  float  v8f;

__device__ __forceinline__ unsigned short f2bf(float f) {
  __bf16 h = (__bf16)f;
  unsigned short u;
  __builtin_memcpy(&u, &h, 2);
  return u;
}

// join two 16B-aligned 8x-bf16 chunks into a WMMA operand (LDS or global)
__device__ __forceinline__ v16bf ld_tile(const unsigned short* p, int off2) {
  v8bf lo = *(const v8bf*)p;
  v8bf hi = *(const v8bf*)(p + off2);
  v16bf r;
#pragma unroll
  for (int e = 0; e < 8; ++e) { r[e] = lo[e]; r[e + 8] = hi[e]; }
  return r;
}

// ---------------- init: zero grid-barrier counter, init const columns ----------------
__global__ void cde_init(unsigned int* cnt, unsigned short* yb0, unsigned short* yb1) {
  int tid = threadIdx.x;
  if (tid == 0) *cnt = 0u;
  for (int i = tid; i < 2 * BB * 32; i += TPB) {
    int bufsel = i >> 11;
    int rem    = i & 2047;
    int b      = rem >> 5;
    int c      = 256 + (rem & 31);
    unsigned short v = (c == 256) ? f2bf(1.0f) : (unsigned short)0;
    (bufsel ? yb1 : yb0)[b * KY + c] = v;
  }
}

// ---------------- pack W_A (+W_B column) into WMMA-B tiled bf16 image ----------------
// tile(nt,kt) = 1024B. First 512B: lane l's low 16B at l*16; second 512B: high 16B.
// Lane l covers n = nt*16+(l&15), k = kt*32 + (l>>4)*16 + 0..15.
__global__ void cde_pack_w(const float* __restrict__ W_A, const float* __restrict__ W_B,
                           unsigned int* __restrict__ Wt) {
  int t    = blockIdx.x * 256 + threadIdx.x;  // output dword id == storage index
  int blk  = t >> 8;                          // (nt,kt) tile
  int w    = t & 255;                         // dword within tile
  int half = w >> 7;
  int lane = (w >> 2) & 31;
  int q    = w & 3;
  int v    = half * 4 + q;                    // lane's dword 0..7
  int nt   = blk / KSTEPS;
  int kt   = blk - nt * KSTEPS;
  int n    = nt * 16 + (lane & 15);
  int k0   = kt * 32 + ((lane >> 4) << 4) + v * 2;
  int h    = n / DPAD;
  int d    = n - h * DPAD;
  float f0 = 0.f, f1 = 0.f;
  if (d < DP) {
    int row = h * DP + d;  // row of W_A; also flat index into W_B (H x (D+1))
    int k1  = k0 + 1;
    f0 = (k0 < 256) ? W_A[row * HH + k0] : ((k0 == 256) ? W_B[row] : 0.f);
    f1 = (k1 < 256) ? W_A[row * HH + k1] : ((k1 == 256) ? W_B[row] : 0.f);
  }
  Wt[t] = (unsigned int)f2bf(f0) | ((unsigned int)f2bf(f1) << 16);
}

// ---------------- grid barrier: monotonic counter ----------------
__device__ __forceinline__ void gridbar(unsigned int* cnt, unsigned int gen) {
  __threadfence();
  __syncthreads();
  if (threadIdx.x == 0) {
    __hip_atomic_fetch_add(cnt, 1u, __ATOMIC_ACQ_REL, __HIP_MEMORY_SCOPE_AGENT);
    while (__hip_atomic_load(cnt, __ATOMIC_RELAXED, __HIP_MEMORY_SCOPE_AGENT) < gen * NWG)
      __builtin_amdgcn_s_sleep(1);
  }
  __syncthreads();
  __threadfence();
}

// ---------------- main persistent kernel: one WG = 4 h's, all 64 batches ----------------
__global__ void __launch_bounds__(TPB, 1)
cde_main(const float* __restrict__ X, const float* __restrict__ W_init,
         const float* __restrict__ b_init, const unsigned int* __restrict__ Wt,
         unsigned short* __restrict__ yb0, unsigned short* __restrict__ yb1,
         unsigned int* __restrict__ cnt, float* __restrict__ out) {
  extern __shared__ __align__(16) char smem[];
  unsigned short* wl  = (unsigned short*)smem;                   // W slice (bf16 tiles)
  float*          uu  = (float*)(smem + WLDS_BYTES);             // u ping-pong
  float*          ysl = (float*)(smem + WLDS_BYTES + UU_BYTES);  // f32 y slice

  const int tid   = threadIdx.x;
  const int lane  = tid & 31;
  const int wave  = tid >> 5;
  const int mt    = wave >> 1;   // M-tile (batch group of 16)
  const int pair  = wave & 1;    // which pair of local h's
  const int hbase = blockIdx.x * HPW;
  const float dt  = 1.0f / (SS - 1);

  // ----- stage this WG's W slice into LDS (once): 11520 uint4, contiguous -----
  {
    const uint4* s4 = (const uint4*)(Wt + (size_t)(hbase * 5 * KSTEPS) * 256);
    uint4* d4 = (uint4*)wl;
#pragma unroll 1
    for (int i = 0; i < WLDS_BYTES / 16 / TPB; ++i)   // 45 iters
      d4[tid + i * TPB] = s4[tid + i * TPB];
  }

  // ----- u buffer for t=1 (buffer 1) -----
#pragma unroll
  for (int i = 0; i < 20; ++i) {
    int idx = tid + i * TPB;
    int b = idx / DPAD;
    int c = idx - b * DPAD;
    float v = 0.f;
    if (c == 0) v = dt;
    else if (c <= DD) v = X[((size_t)b * SS + 1) * DD + (c - 1)];
    uu[UU_FLOATS + idx] = v;
  }

  // ----- y0 = X[:,0,:] @ W_init.T + b_init (one thread per (b, h_local)) -----
  {
    int b = tid & 63;
    int hl = tid >> 6;
    int h = hbase + hl;
    float acc = b_init[h];
    const float* xr = X + (size_t)b * SS * DD;  // t = 0
    const float* wr = W_init + h * DD;
#pragma unroll 8
    for (int j = 0; j < DD; ++j) acc += xr[j] * wr[j];
    ysl[b * HPW + hl] = acc;
    out[((size_t)b * SS + 0) * HH + h] = acc;
    yb0[b * KY + h] = f2bf(acc);
  }
  gridbar(cnt, 1u);

  // ----- sequential scan over timesteps -----
  for (int t = 1; t < SS; ++t) {
    const unsigned short* src = (t & 1) ? yb0 : yb1;  // y_{t-1}
    unsigned short*       dst = (t & 1) ? yb1 : yb0;  // y_t

    // A-tiles (this wave's 16 batch rows of y_{t-1}) straight from global -> regs
    v16bf a[9];
    {
      const unsigned short* yr =
          src + (mt * 16 + (lane & 15)) * KY + ((lane >> 4) << 4);
#pragma unroll
      for (int kt = 0; kt < 9; ++kt) a[kt] = ld_tile(yr + kt * 32, 8);
    }

    const float* uuc = uu + (size_t)(t & 1) * UU_FLOATS;        // current u_t
    float*       uun = uu + (size_t)((t + 1) & 1) * UU_FLOATS;  // next u_{t+1}

    // prefetch u_{t+1} into the other LDS buffer (off the critical path)
    if (t + 1 < SS) {
#pragma unroll
      for (int i = 0; i < 20; ++i) {
        int idx = tid + i * TPB;
        int b = idx / DPAD;
        int c = idx - b * DPAD;
        float v = 0.f;
        if (c == 0) v = dt;
        else if (c <= DD) v = X[((size_t)b * SS + (t + 1)) * DD + (c - 1)];
        uun[idx] = v;
      }
    }

    const int bofs = mt * 16 + ((lane >> 4) << 3);
    const int dcol = lane & 15;

    // two h's per wave; 5 d-tiles x 9 k-steps of WMMA each, fused d-contraction
#pragma unroll 1
    for (int uidx = 0; uidx < 2; ++uidx) {
      int hl = pair * 2 + uidx;
      int h  = hbase + hl;
      float part[8] = {0.f, 0.f, 0.f, 0.f, 0.f, 0.f, 0.f, 0.f};
#pragma unroll 1
      for (int t5 = 0; t5 < 5; ++t5) {
        // contraction weights u[b(r,lane), d(t5,lane)]
        float wgt8[8];
#pragma unroll
        for (int r = 0; r < 8; ++r)
          wgt8[r] = uuc[(bofs + r) * DPAD + t5 * 16 + dcol];

        const unsigned short* bp =
            wl + (size_t)((hl * 5 + t5) * KSTEPS) * 512 + lane * 8;  // ushort units
        v8f acc0 = {}; v8f acc1 = {};
        v16bf cur = ld_tile(bp, 256);
#pragma unroll
        for (int kt = 0; kt < 9; ++kt) {
          v16bf nxt;
          if (kt < 8) nxt = ld_tile(bp + (kt + 1) * 512, 256);  // double-buffer
          if (kt & 1)
            acc1 = __builtin_amdgcn_wmma_f32_16x16x32_bf16(false, a[kt], false, cur,
                                                           (short)0, acc1, false, false);
          else
            acc0 = __builtin_amdgcn_wmma_f32_16x16x32_bf16(false, a[kt], false, cur,
                                                           (short)0, acc0, false, false);
          if (kt < 8) cur = nxt;
        }
#pragma unroll
        for (int r = 0; r < 8; ++r) part[r] += (acc0[r] + acc1[r]) * wgt8[r];
      }
      // reduce over the 16 d-columns (within each 16-lane half)
      float red[8];
#pragma unroll
      for (int r = 0; r < 8; ++r) {
        float s = part[r];
        s += __shfl_xor(s, 1, 32);
        s += __shfl_xor(s, 2, 32);
        s += __shfl_xor(s, 4, 32);
        s += __shfl_xor(s, 8, 32);
        red[r] = s;
      }
      // gather the 16 batch scalars onto lanes 0..15 (b = lane)
      float yv = 0.f;
#pragma unroll
      for (int r = 0; r < 8; ++r) {
        float other = __shfl_xor(red[r], 16, 32);
        if ((lane & 7) == r) yv = (lane & 8) ? other : red[r];
      }
      if (lane < 16) {
        int b = mt * 16 + lane;
        float ynew = ysl[b * HPW + hl] + yv;  // A·u and W_B·u both inside yv
        ysl[b * HPW + hl] = ynew;
        out[((size_t)b * SS + t) * HH + h] = ynew;
        dst[b * KY + h] = f2bf(ynew);
      }
    }
    gridbar(cnt, (unsigned int)(t + 1));
  }
}

// ---------------- launch ----------------
extern "C" void kernel_launch(void* const* d_in, const int* in_sizes, int n_in,
                              void* d_out, int out_size, void* d_ws, size_t ws_size,
                              hipStream_t stream) {
  (void)in_sizes; (void)n_in; (void)out_size; (void)ws_size;
  const float* X      = (const float*)d_in[0];
  const float* W_init = (const float*)d_in[1];
  const float* b_init = (const float*)d_in[2];
  const float* W_A    = (const float*)d_in[3];
  const float* W_B    = (const float*)d_in[4];
  float* out = (float*)d_out;

  char* ws = (char*)d_ws;
  const size_t wtBytes = (size_t)NTILES * KSTEPS * 1024;  // 11,796,480
  unsigned int*   Wt  = (unsigned int*)ws;
  unsigned short* yb0 = (unsigned short*)(ws + wtBytes);
  unsigned short* yb1 = yb0 + BB * KY;
  unsigned int*   cnt = (unsigned int*)(ws + wtBytes + (size_t)2 * BB * KY * 2);

  (void)hipFuncSetAttribute((const void*)cde_main,
                            hipFuncAttributeMaxDynamicSharedMemorySize, LDS_BYTES);

  cde_init<<<1, TPB, 0, stream>>>(cnt, yb0, yb1);
  cde_pack_w<<<NTILES * KSTEPS, 256, 0, stream>>>(W_A, W_B, Wt);
  cde_main<<<NWG, TPB, LDS_BYTES, stream>>>(X, W_init, b_init, Wt, yb0, yb1, cnt, out);
}
